// WaveNetModel_12094627906063
// MI455X (gfx1250) — matrix operand order repaired and checked by reading, not compiled, not run on hardware
//
#include <hip/hip_runtime.h>
#include <hip/hip_bf16.h>

// ---------------------------------------------------------------------------
// WaveNet fused kernel for gfx1250 (MI455X).
//   B=8, T=32768, CIN=1, R=64, S=128, NB=3, NL=8, K=2, NBLK=24
//   dilations = 1,2,4,...,128 repeated 3x; total receptive field = 765.
// One workgroup = 256 threads (8 wave32) owns a 1024-column h-tile in LDS and
// emits 256 valid timesteps. All GEMMs via v_wmma_f32_16x16x32_f16.
// ---------------------------------------------------------------------------

typedef __attribute__((ext_vector_type(16))) _Float16 v16h;
typedef __attribute__((ext_vector_type(8)))  _Float16 v8h;
typedef __attribute__((ext_vector_type(8)))  float    v8f;
typedef __attribute__((ext_vector_type(4)))  float    v4f;

#define NBLK 24
#define RCH  64
#define SCH  128
#define TDIM 32768
#define BDIM 8
#define CTILE 1024          // buffer columns per workgroup
#define TT    256           // valid output columns per workgroup
#define HALO  768           // >= 765
#define HSTR  72            // padded column stride (halves) for 64-row buffers
#define SSTR  136           // padded column stride (halves) for 128-row skip
#define OSTR  132           // padded column stride (floats) for 128-row out1
#define TILE_HALVES 512     // 16x32 f16 A-fragment
#define BLK_TILES   56      // 8(c0)+8(c1)+8(f)+8(g)+8(res)+16(skip)
#define W1_BASE     (NBLK * BLK_TILES)          // 1344
#define TOTAL_TILES (W1_BASE + 32)              // + W1 (128x128 -> 8x4 tiles)
#define ZPAD_OFF    (2 * CTILE * HSTR + 8 * 16 * HSTR)  // zero pad column (72 halves)

static __device__ __forceinline__ v8f wmma_f(v16h a, v16h b, v8f c) {
  return __builtin_amdgcn_wmma_f32_16x16x32_f16(false, a, false, b,
                                                (short)0, c, false, false);
}

static __device__ __forceinline__ v16h cat16(v8h lo, v8h hi) {
  v16h r;
#pragma unroll
  for (int i = 0; i < 8; ++i) { r[i] = lo[i]; r[i + 8] = hi[i]; }
  return r;
}

// A-fragment load: pre-packed, lane-contiguous (16 halves / lane / tile).
static __device__ __forceinline__ v16h ldA(const _Float16* __restrict__ wpk,
                                           int tile, int lane) {
  const _Float16* p = wpk + ((size_t)tile << 9) + (lane << 4);
  return cat16(*(const v8h*)p, *(const v8h*)(p + 8));
}

// B-fragment load from LDS column-major buffer (16B-aligned offsets).
static __device__ __forceinline__ v16h ldB(const _Float16* p) {
  return cat16(*(const v8h*)p, *(const v8h*)(p + 8));
}

// Native transcendentals on gfx1250 where available.
static __device__ __forceinline__ float fast_tanh(float x) {
#if __has_builtin(__builtin_amdgcn_tanhf)
  return __builtin_amdgcn_tanhf(x);
#else
  return tanhf(x);
#endif
}
static __device__ __forceinline__ float fast_sigmoid(float x) {
#if __has_builtin(__builtin_amdgcn_rcpf)
  return __builtin_amdgcn_rcpf(1.f + __expf(-x));
#else
  return 1.f / (1.f + __expf(-x));
#endif
}

// ---------------------------------------------------------------------------
// Weight pre-pack: f32 -> f16 A-fragment layout (16x32 tile, contiguous k order matching the LDS B loads):
//   lane = m_local + 16*g ; half q in [0,16) -> k_local = q + 16*g
// ---------------------------------------------------------------------------
__global__ void wavenet_pack(const float* __restrict__ wc,
                             const float* __restrict__ wf,
                             const float* __restrict__ wg,
                             const float* __restrict__ wr,
                             const float* __restrict__ ws,
                             const float* __restrict__ w1,
                             _Float16* __restrict__ out) {
  int idx = blockIdx.x * 256 + threadIdx.x;
  if (idx >= TOTAL_TILES * TILE_HALVES) return;
  int tile   = idx >> 9;
  int within = idx & 511;
  int lane = within >> 4;
  int q    = within & 15;
  int g  = lane >> 4;
  int ml = lane & 15;
  int kl = q + 16 * g;
  float val;
  if (tile < W1_BASE) {
    int blk = tile / BLK_TILES;
    int r   = tile % BLK_TILES;
    if (r < 16) {                       // causal conv taps: w_causal[blk][o][i][k]
      int tap = r >> 3; int rr = r & 7;
      int mt = rr >> 1, kt = rr & 1;
      int m = mt * 16 + ml, k = kt * 32 + kl;
      val = wc[(((size_t)blk * 64 + m) * 64 + k) * 2 + tap];
    } else if (r < 40) {                // filter / gate / res (64x64)
      int which = (r - 16) >> 3; int rr = (r - 16) & 7;
      int mt = rr >> 1, kt = rr & 1;
      int m = mt * 16 + ml, k = kt * 32 + kl;
      const float* W = (which == 0) ? wf : ((which == 1) ? wg : wr);
      val = W[(size_t)blk * 4096 + m * 64 + k];
    } else {                            // skip (128x64)
      int rr = r - 40;
      int mt = rr >> 1, kt = rr & 1;
      int m = mt * 16 + ml, k = kt * 32 + kl;
      val = ws[(size_t)blk * 8192 + m * 64 + k];
    }
  } else {                              // w_out1 (128x128)
    int r = tile - W1_BASE;
    int mt = r >> 2, kt = r & 3;
    int m = mt * 16 + ml, k = kt * 32 + kl;
    val = w1[m * 128 + k];
  }
  out[idx] = (_Float16)val;
}

// ---------------------------------------------------------------------------
// Fused residual stack + head.
// ---------------------------------------------------------------------------
__global__ __launch_bounds__(256)
void wavenet_fused(const float* __restrict__ x,
                   const float* __restrict__ w_in,  const float* __restrict__ b_in,
                   const float* __restrict__ b_causal,
                   const float* __restrict__ b_filter,
                   const float* __restrict__ b_gate,
                   const float* __restrict__ b_res,
                   const float* __restrict__ b_skip,
                   const float* __restrict__ b_out1,
                   const float* __restrict__ w_out2,
                   const float* __restrict__ b_out2,
                   const _Float16* __restrict__ wpk,
                   float* __restrict__ out) {
  extern __shared__ _Float16 smem[];
  _Float16* h_lds = smem;                         // [CTILE][HSTR] f16
  _Float16* a_lds = smem + CTILE * HSTR;          // [CTILE][HSTR] f16

  const int tid  = threadIdx.x;
  const int lane = tid & 31;
  const int wave = tid >> 5;
  const int g    = lane >> 4;
  const int ln   = lane & 15;
  _Float16* zscr = smem + 2 * CTILE * HSTR + wave * 16 * HSTR;  // per-wave z
  const _Float16* zpad = smem + ZPAD_OFF;         // always-zero pad column

  const int b    = blockIdx.x >> 7;               // batch (8)
  const int seg  = blockIdx.x & 127;              // 128 segments of 256
  const int t0   = seg << 8;
  const int tseg = t0 - HALO;                     // global time of buffer col 0

  // ---- zero pad column + input 1x1 conv into h tile --------------------
  if (tid < HSTR) smem[ZPAD_OFF + tid] = (_Float16)0.f;
  for (int c = tid; c < CTILE; c += 256) {
    int t = tseg + c;
    float xv = (t >= 0 && t < TDIM) ? x[(size_t)b * TDIM + t] : 0.f;
    _Float16* hc = &h_lds[c * HSTR];
    for (int r = 0; r < RCH; ++r)
      hc[r] = (_Float16)(w_in[r] * xv + b_in[r]);
  }

  v8f sacc[2][8];
#pragma unroll
  for (int j = 0; j < 2; ++j)
#pragma unroll
    for (int mt = 0; mt < 8; ++mt)
#pragma unroll
      for (int v = 0; v < 8; ++v) sacc[j][mt][v] = 0.f;

  __syncthreads();

  // ---- 24 residual blocks ---------------------------------------------
#pragma unroll 1
  for (int blk = 0; blk < NBLK; ++blk) {
    const int d = 1 << (blk & 7);
    const int wb = blk * BLK_TILES;

    // Phase A: z = Wc0*h[t-d] + Wc1*h[t] + bc ; a = tanh(f)*sig(g)
#pragma unroll 1
    for (int it = 0; it < 8; ++it) {
      const int col = (wave + 8 * it) * 16 + ln;
      v16h b1k0 = ldB(&h_lds[col * HSTR + 16 * g]);
      v16h b1k1 = ldB(&h_lds[col * HSTR + 32 + 16 * g]);
      // causal zero padding via address redirect to the zero column
      int  cs  = col - d;
      bool zz  = (tseg + cs) < 0;
      int  csc = cs < 0 ? 0 : cs;
      const _Float16* p0 = zz ? zpad : &h_lds[csc * HSTR];
      v16h b0k0 = ldB(p0 + 16 * g);
      v16h b0k1 = ldB(p0 + 32 + 16 * g);
#pragma unroll
      for (int mt = 0; mt < 4; ++mt) {
        const int row = mt * 16 + 8 * g;
        v8f acc;
#pragma unroll
        for (int v = 0; v < 8; ++v) acc[v] = b_causal[blk * 64 + row + v];
        acc = wmma_f(ldA(wpk, wb + mt * 2 + 0, lane), b0k0, acc);
        acc = wmma_f(ldA(wpk, wb + mt * 2 + 1, lane), b0k1, acc);
        acc = wmma_f(ldA(wpk, wb + 8 + mt * 2 + 0, lane), b1k0, acc);
        acc = wmma_f(ldA(wpk, wb + 8 + mt * 2 + 1, lane), b1k1, acc);
        v8h zst;
#pragma unroll
        for (int v = 0; v < 8; ++v) zst[v] = (_Float16)acc[v];
        *(v8h*)&zscr[ln * HSTR + row] = zst;
      }
      v16h bz0 = ldB(&zscr[ln * HSTR + 16 * g]);
      v16h bz1 = ldB(&zscr[ln * HSTR + 32 + 16 * g]);
#pragma unroll
      for (int mt = 0; mt < 4; ++mt) {
        const int row = mt * 16 + 8 * g;
        v8f fac, gac;
#pragma unroll
        for (int v = 0; v < 8; ++v) {
          fac[v] = b_filter[blk * 64 + row + v];
          gac[v] = b_gate[blk * 64 + row + v];
        }
        fac = wmma_f(ldA(wpk, wb + 16 + mt * 2 + 0, lane), bz0, fac);
        fac = wmma_f(ldA(wpk, wb + 16 + mt * 2 + 1, lane), bz1, fac);
        gac = wmma_f(ldA(wpk, wb + 24 + mt * 2 + 0, lane), bz0, gac);
        gac = wmma_f(ldA(wpk, wb + 24 + mt * 2 + 1, lane), bz1, gac);
        v8h ast;
#pragma unroll
        for (int v = 0; v < 8; ++v) {
          float tv = fast_tanh(fac[v]);
          float sv = fast_sigmoid(gac[v]);
          ast[v] = (_Float16)(tv * sv);
        }
        *(v8h*)&a_lds[col * HSTR + row] = ast;
      }
    }

    // skip += Wskip * a + b_skip  (valid tiles only; register accumulators)
#pragma unroll
    for (int j = 0; j < 2; ++j) {
      const int col = (48 + wave + 8 * j) * 16 + ln;
      v16h ba0 = ldB(&a_lds[col * HSTR + 16 * g]);
      v16h ba1 = ldB(&a_lds[col * HSTR + 32 + 16 * g]);
#pragma unroll
      for (int mt = 0; mt < 8; ++mt) {
        const int row = mt * 16 + 8 * g;
        v8f acc = sacc[j][mt];
#pragma unroll
        for (int v = 0; v < 8; ++v) acc[v] += b_skip[blk * 128 + row + v];
        acc = wmma_f(ldA(wpk, wb + 40 + mt * 2 + 0, lane), ba0, acc);
        acc = wmma_f(ldA(wpk, wb + 40 + mt * 2 + 1, lane), ba1, acc);
        sacc[j][mt] = acc;
      }
    }

    __syncthreads();

    // Phase B: h += Wres * a + b_res (in place; each wave owns its columns)
#pragma unroll 1
    for (int it = 0; it < 8; ++it) {
      const int col = (wave + 8 * it) * 16 + ln;
      v16h ba0 = ldB(&a_lds[col * HSTR + 16 * g]);
      v16h ba1 = ldB(&a_lds[col * HSTR + 32 + 16 * g]);
#pragma unroll
      for (int mt = 0; mt < 4; ++mt) {
        const int row = mt * 16 + 8 * g;
        v8h hold = *(const v8h*)&h_lds[col * HSTR + row];
        v8f acc;
#pragma unroll
        for (int v = 0; v < 8; ++v)
          acc[v] = (float)hold[v] + b_res[blk * 64 + row + v];
        acc = wmma_f(ldA(wpk, wb + 32 + mt * 2 + 0, lane), ba0, acc);
        acc = wmma_f(ldA(wpk, wb + 32 + mt * 2 + 1, lane), ba1, acc);
        v8h hnew;
#pragma unroll
        for (int v = 0; v < 8; ++v) hnew[v] = (_Float16)acc[v];
        *(v8h*)&h_lds[col * HSTR + row] = hnew;
      }
    }
    __syncthreads();
  }

  // ---- head: relu(skip) -> W1+b1 -> relu -> W2+b2 ----------------------
  // reuse h region as skip buffer [256][SSTR] f16
#pragma unroll
  for (int j = 0; j < 2; ++j) {
    const int hc = (wave + 8 * j) * 16 + ln;
#pragma unroll
    for (int mt = 0; mt < 8; ++mt) {
      const int row = mt * 16 + 8 * g;
      v8f s = sacc[j][mt];
      v8h st;
#pragma unroll
      for (int v = 0; v < 8; ++v) {
        float r = s[v] > 0.f ? s[v] : 0.f;
        st[v] = (_Float16)r;
      }
      *(v8h*)&h_lds[hc * SSTR + row] = st;
    }
  }
  __syncthreads();

  float* out1_lds = (float*)(smem + CTILE * HSTR);  // [256][OSTR] f32
#pragma unroll
  for (int j = 0; j < 2; ++j) {
    const int hc = (wave + 8 * j) * 16 + ln;
    v16h bs0 = ldB(&h_lds[hc * SSTR + 0  + 16 * g]);
    v16h bs1 = ldB(&h_lds[hc * SSTR + 32 + 16 * g]);
    v16h bs2 = ldB(&h_lds[hc * SSTR + 64 + 16 * g]);
    v16h bs3 = ldB(&h_lds[hc * SSTR + 96 + 16 * g]);
#pragma unroll
    for (int mt = 0; mt < 8; ++mt) {
      const int row = mt * 16 + 8 * g;
      v8f acc;
#pragma unroll
      for (int v = 0; v < 8; ++v) acc[v] = b_out1[row + v];
      acc = wmma_f(ldA(wpk, W1_BASE + mt * 4 + 0, lane), bs0, acc);
      acc = wmma_f(ldA(wpk, W1_BASE + mt * 4 + 1, lane), bs1, acc);
      acc = wmma_f(ldA(wpk, W1_BASE + mt * 4 + 2, lane), bs2, acc);
      acc = wmma_f(ldA(wpk, W1_BASE + mt * 4 + 3, lane), bs3, acc);
      v4f lo, hi;
#pragma unroll
      for (int v = 0; v < 4; ++v) {
        lo[v] = acc[v]     > 0.f ? acc[v]     : 0.f;
        hi[v] = acc[v + 4] > 0.f ? acc[v + 4] : 0.f;
      }
      *(v4f*)&out1_lds[hc * OSTR + row]     = lo;
      *(v4f*)&out1_lds[hc * OSTR + row + 4] = hi;
    }
  }
  __syncthreads();

  // final 1x128 dot per output column
  {
    const int colT = tid;             // 0..255
    float sum = b_out2[0];
    const float* o1 = &out1_lds[colT * OSTR];
#pragma unroll 8
    for (int r = 0; r < SCH; ++r) sum += w_out2[r] * o1[r];
    out[(size_t)b * TDIM + t0 + colT] = sum;
  }
}

// ---------------------------------------------------------------------------
extern "C" void kernel_launch(void* const* d_in, const int* in_sizes, int n_in,
                              void* d_out, int out_size, void* d_ws, size_t ws_size,
                              hipStream_t stream) {
  (void)in_sizes; (void)n_in; (void)out_size; (void)ws_size;
  const float* x        = (const float*)d_in[0];
  const float* w_in     = (const float*)d_in[1];
  const float* b_in     = (const float*)d_in[2];
  const float* w_causal = (const float*)d_in[3];
  const float* b_causal = (const float*)d_in[4];
  const float* w_filter = (const float*)d_in[5];
  const float* b_filter = (const float*)d_in[6];
  const float* w_gate   = (const float*)d_in[7];
  const float* b_gate   = (const float*)d_in[8];
  const float* w_res    = (const float*)d_in[9];
  const float* b_res    = (const float*)d_in[10];
  const float* w_skip   = (const float*)d_in[11];
  const float* b_skip   = (const float*)d_in[12];
  const float* w_out1   = (const float*)d_in[13];
  const float* b_out1   = (const float*)d_in[14];
  const float* w_out2   = (const float*)d_in[15];
  const float* b_out2   = (const float*)d_in[16];
  _Float16* wpk = (_Float16*)d_ws;

  // pack all weights into CDNA5 A-fragment layout (f16), once per launch
  const int packN = TOTAL_TILES * TILE_HALVES;
  wavenet_pack<<<packN / 256, 256, 0, stream>>>(w_causal, w_filter, w_gate,
                                                w_res, w_skip, w_out1, wpk);

  // fused stack + head: ~306KB dynamic LDS (h + a + z-scratch + zero pad)
  const size_t ldsBytes =
      (size_t)(2 * CTILE * HSTR + 8 * 16 * HSTR + HSTR) * sizeof(_Float16);
  wavenet_fused<<<dim3(BDIM * 128), dim3(256), ldsBytes, stream>>>(
      x, w_in, b_in, b_causal, b_filter, b_gate, b_res, b_skip,
      b_out1, w_out2, b_out2, wpk, (float*)d_out);
}
